// GCNModel_pooled_89541478187032
// MI455X (gfx1250) — compile-verified, matrix operand order checked by reading, not run on hardware
//
#include <hip/hip_runtime.h>
#include <hip/hip_bf16.h>
#include <math.h>

// ---------------------------------------------------------------- constants
#define BS       4
#define NNODES   10000
#define NFEAT    64
#define HID      128
#define NLAYERS  4
#define HIDFC    512
#define NCLASSES 10
#define NEDGES   160000
#define NTOT     (BS * NNODES)        // 40000 rows
#define ECSR     (NEDGES + NNODES)    // edges + self loops

typedef __attribute__((ext_vector_type(2))) float v2f;
typedef __attribute__((ext_vector_type(8))) float v8f;

#if defined(__has_builtin)
#if __has_builtin(__builtin_amdgcn_wmma_f32_16x16x4_f32)
#define HAVE_F32_WMMA 1
#endif
#endif
#ifndef HAVE_F32_WMMA
#define HAVE_F32_WMMA 0
typedef __attribute__((ext_vector_type(16))) __bf16 v16bf;
#endif

// ---------------------------------------------------------------- CSR build
__global__ void k_deg_init(int* deg) {
    int i = blockIdx.x * blockDim.x + threadIdx.x;
    if (i < NNODES) deg[i] = 1;   // self loop
}

__global__ void k_deg_count(int* deg, const int* __restrict__ dst) {
    int e = blockIdx.x * blockDim.x + threadIdx.x;
    if (e < NEDGES) atomicAdd(&deg[dst[e]], 1);
}

__global__ void k_dinv(const int* __restrict__ deg, float* __restrict__ dinv) {
    int i = blockIdx.x * blockDim.x + threadIdx.x;
    if (i < NNODES) dinv[i] = rsqrtf((float)deg[i]);   // deg >= 1 always
}

// single-block exclusive scan of degrees -> row_ptr (n = 10000, 1024 threads)
__global__ void k_scan(const int* __restrict__ deg, int* __restrict__ rp) {
    __shared__ int part[1024];
    int t = threadIdx.x;
    const int per = (NNODES + 1023) / 1024;
    int s0 = t * per;
    int s1 = min(s0 + per, NNODES);
    int s = 0;
    for (int i = s0; i < s1; ++i) s += deg[i];
    part[t] = s;
    __syncthreads();
    for (int o = 1; o < 1024; o <<= 1) {
        int v = (t >= o) ? part[t - o] : 0;
        __syncthreads();
        part[t] += v;
        __syncthreads();
    }
    int run = (t == 0) ? 0 : part[t - 1];
    for (int i = s0; i < s1; ++i) { rp[i] = run; run += deg[i]; }
    if (t == 1023) rp[NNODES] = part[1023];
}

__global__ void k_copy_int(const int* __restrict__ a, int* __restrict__ b) {
    int i = blockIdx.x * blockDim.x + threadIdx.x;
    if (i < NNODES) b[i] = a[i];
}

__global__ void k_csr_fill(const int* __restrict__ src, const int* __restrict__ dst,
                           const float* __restrict__ dinv, int* __restrict__ fcnt,
                           int* __restrict__ csrc, float* __restrict__ cw) {
    int t = blockIdx.x * blockDim.x + threadIdx.x;
    if (t >= ECSR) return;
    int s, d;
    if (t < NEDGES) { s = src[t]; d = dst[t]; }
    else            { s = d = t - NEDGES; }          // self loop
    int slot = atomicAdd(&fcnt[d], 1);
    csrc[slot] = s;
    cw[slot]   = dinv[s] * dinv[d];
}

// ---------------------------------------------------------------- WMMA GEMM
// Out[M,128] = A[M,K] @ W[K,128].  One block = one 16-row M tile; 8 waves,
// wave w owns N-tile w (16 cols).  f32 path uses V_WMMA_F32_16X16X4_F32.
// A-frag layout (ISA 7.12.2, 32-bit A 16x4): lane m = lane&15,
// k = k0 + 2*(lane>>4) + j (j = vector component).  B mirrors with n = lane&15.
__global__ void k_gemm_wmma(const float* __restrict__ A, const float* __restrict__ W,
                            float* __restrict__ Out, int K) {
    const int N = HID;
    int wave = threadIdx.x >> 5;         // 0..7 -> N tile
    int lane = threadIdx.x & 31;
    int half = lane >> 4;
    int mn   = lane & 15;
    size_t rowBase = (size_t)blockIdx.x * 16;
    int    colBase = wave * 16;
    v8f c = {};
#if HAVE_F32_WMMA
    for (int k0 = 0; k0 < K; k0 += 4) {
        int ka = k0 + 2 * half;
        v2f a, b;
        a.x = A[(rowBase + mn) * K + ka];
        a.y = A[(rowBase + mn) * K + ka + 1];
        b.x = W[(size_t)ka * N + colBase + mn];
        b.y = W[(size_t)(ka + 1) * N + colBase + mn];
        c = __builtin_amdgcn_wmma_f32_16x16x4_f32(false, a, false, b,
                                                  (short)0, c, false, false);
    }
#else
    for (int k0 = 0; k0 < K; k0 += 32) {
        v16bf a, b;
#pragma unroll
        for (int e = 0; e < 16; ++e) {
            int k = k0 + e + 8 * half + ((e >= 8) ? 8 : 0);
            a[e] = (__bf16)A[(rowBase + mn) * K + k];
            b[e] = (__bf16)W[(size_t)k * N + colBase + mn];
        }
        c = __builtin_amdgcn_wmma_f32_16x16x32_bf16(false, a, false, b,
                                                    (short)0, c, false, false);
    }
#endif
#pragma unroll
    for (int j = 0; j < 8; ++j)
        Out[(rowBase + j + 8 * half) * N + colBase + mn] = c[j];
}

// ---------------------------------------------------------------- aggregation
// hout[b,i,:] = relu( sum_{s in csr(i)} cw[s] * hw[b, csrc[s], :] + bias )
// One block = (node, batch); 128 threads = feature lanes. Coalesced 512B reads.
__global__ void k_agg(const float* __restrict__ hw, const int* __restrict__ rp,
                      const int* __restrict__ csrc, const float* __restrict__ cw,
                      const float* __restrict__ bias, float* __restrict__ hout) {
    int i = blockIdx.x;
    int b = blockIdx.y;
    int f = threadIdx.x;
    int s0 = rp[i], s1 = rp[i + 1];
    const size_t bBase = (size_t)b * NNODES;
    float acc = 0.f;
    for (int s = s0; s < s1; ++s) {
        int   src = csrc[s];
        float w   = cw[s];
        if (s + 1 < s1)   // gfx1250 global_prefetch_b8 of next gather row
            __builtin_prefetch(&hw[(bBase + csrc[s + 1]) * HID + f], 0, 1);
        acc += w * hw[(bBase + src) * HID + f];
    }
    hout[(bBase + i) * HID + f] = fmaxf(acc + bias[f], 0.f);
}

// ---------------------------------------------------------------- mean pool
// pooled[b, ch*NLAYERS + layer] = mean_i h[b,i,ch]  (JK interleave layout)
__global__ void k_pool(const float* __restrict__ h, float* __restrict__ pooled,
                       int layer) {
    int b  = blockIdx.x / HID;
    int ch = blockIdx.x % HID;
    int t  = threadIdx.x;
    __shared__ float red[256];
    float s = 0.f;
    for (int i = t; i < NNODES; i += 256)
        s += h[((size_t)b * NNODES + i) * HID + ch];
    red[t] = s;
    __syncthreads();
    for (int o = 128; o > 0; o >>= 1) {
        if (t < o) red[t] += red[t + o];
        __syncthreads();
    }
    if (t == 0)
        pooled[b * (NLAYERS * HID) + ch * NLAYERS + layer] = red[0] * (1.0f / NNODES);
}

// ---------------------------------------------------------------- MLP head
__global__ void k_head(const float* __restrict__ pooled,
                       const float* __restrict__ w1, const float* __restrict__ b1,
                       const float* __restrict__ w2, const float* __restrict__ b2,
                       float* __restrict__ out) {
    __shared__ float P[BS * HIDFC];
    __shared__ float H1[BS * HIDFC];
    __shared__ float LG[BS * NCLASSES];
    int t = threadIdx.x;   // 512 threads
    for (int i = t; i < BS * HIDFC; i += HIDFC) P[i] = pooled[i];
    __syncthreads();
    float acc[BS];
#pragma unroll
    for (int b = 0; b < BS; ++b) acc[b] = b1[t];
    for (int k = 0; k < HIDFC; ++k) {
        float w = w1[(size_t)k * HIDFC + t];
#pragma unroll
        for (int b = 0; b < BS; ++b) acc[b] += P[b * HIDFC + k] * w;
    }
#pragma unroll
    for (int b = 0; b < BS; ++b) H1[b * HIDFC + t] = fmaxf(acc[b], 0.f);
    __syncthreads();
    if (t < BS * NCLASSES) {
        int b = t / NCLASSES, c = t % NCLASSES;
        float a = b2[c];
        for (int k = 0; k < HIDFC; ++k) a += H1[b * HIDFC + k] * w2[k * NCLASSES + c];
        LG[t] = a;
    }
    __syncthreads();
    if (t < BS) {
        float mx = -INFINITY;
        for (int c = 0; c < NCLASSES; ++c) mx = fmaxf(mx, LG[t * NCLASSES + c]);
        float s = 0.f;
        for (int c = 0; c < NCLASSES; ++c) s += expf(LG[t * NCLASSES + c] - mx);
        float lse = mx + logf(s);
        for (int c = 0; c < NCLASSES; ++c) out[t * NCLASSES + c] = LG[t * NCLASSES + c] - lse;
    }
}

// ---------------------------------------------------------------- launcher
extern "C" void kernel_launch(void* const* d_in, const int* in_sizes, int n_in,
                              void* d_out, int out_size, void* d_ws, size_t ws_size,
                              hipStream_t stream) {
    const float* x      = (const float*)d_in[0];
    const int*   ei     = (const int*)  d_in[1];
    // d_in[2] (batch) is implied by row-major [BS, NNODES] layout
    const float* W1     = (const float*)d_in[3];
    const float* b1     = (const float*)d_in[4];
    const float* Ws     = (const float*)d_in[5];
    const float* bconvs = (const float*)d_in[6];
    const float* l1w    = (const float*)d_in[7];
    const float* l1b    = (const float*)d_in[8];
    const float* l2w    = (const float*)d_in[9];
    const float* l2b    = (const float*)d_in[10];
    float* out = (float*)d_out;

    char*  ws  = (char*)d_ws;
    size_t off = 0;
    auto walloc = [&](size_t bytes) -> void* {
        void* p = ws + off;
        off = (off + bytes + 255) & ~(size_t)255;
        return p;
    };
    float* hA     = (float*)walloc(sizeof(float) * (size_t)NTOT * HID);  // 20.5 MB
    float* hB     = (float*)walloc(sizeof(float) * (size_t)NTOT * HID);  // 20.5 MB
    int*   deg    = (int*)  walloc(sizeof(int)   * NNODES);
    float* dinv   = (float*)walloc(sizeof(float) * NNODES);
    int*   rp     = (int*)  walloc(sizeof(int)   * (NNODES + 1));
    int*   fcnt   = (int*)  walloc(sizeof(int)   * NNODES);
    int*   csrc   = (int*)  walloc(sizeof(int)   * ECSR);
    float* cw     = (float*)walloc(sizeof(float) * ECSR);
    float* pooled = (float*)walloc(sizeof(float) * BS * NLAYERS * HID);
    (void)ws_size; (void)in_sizes; (void)n_in; (void)out_size;

    const int* srcE = ei;            // edge_index[0]
    const int* dstE = ei + NEDGES;   // edge_index[1]

    // CSR build (D^-1/2 (A+I) D^-1/2 weights)
    k_deg_init <<<(NNODES + 255) / 256, 256, 0, stream>>>(deg);
    k_deg_count<<<(NEDGES + 255) / 256, 256, 0, stream>>>(deg, dstE);
    k_dinv     <<<(NNODES + 255) / 256, 256, 0, stream>>>(deg, dinv);
    k_scan     <<<1, 1024, 0, stream>>>(deg, rp);
    k_copy_int <<<(NNODES + 255) / 256, 256, 0, stream>>>(rp, fcnt);
    k_csr_fill <<<(ECSR + 255) / 256, 256, 0, stream>>>(srcE, dstE, dinv, fcnt, csrc, cw);

    // layer 0: x [40000,64] @ W1 -> hB ; aggregate -> hA ; pool
    k_gemm_wmma<<<NTOT / 16, 256, 0, stream>>>(x, W1, hB, NFEAT);
    k_agg      <<<dim3(NNODES, BS), HID, 0, stream>>>(hB, rp, csrc, cw, b1, hA);
    k_pool     <<<BS * HID, 256, 0, stream>>>(hA, pooled, 0);

    for (int l = 1; l < NLAYERS; ++l) {
        k_gemm_wmma<<<NTOT / 16, 256, 0, stream>>>(hA, Ws + (size_t)(l - 1) * HID * HID, hB, HID);
        k_agg      <<<dim3(NNODES, BS), HID, 0, stream>>>(hB, rp, csrc, cw,
                                                          bconvs + (l - 1) * HID, hA);
        k_pool     <<<BS * HID, 256, 0, stream>>>(hA, pooled, l);
    }

    k_head<<<1, HIDFC, 0, stream>>>(pooled, l1w, l1b, l2w, l2b, out);
}
